// SimpleSelfAttention_16492674417185
// MI455X (gfx1250) — compile-verified, matrix operand order checked by reading, not run on hardware
//
#include <hip/hip_runtime.h>
#include <hip/hip_bf16.h>

// ---------------------------------------------------------------------------
// Self-attention for MI455X (gfx1250, wave32) built on v_wmma_f32_16x16x32_bf16
//   Kernel 0: one-shot f32 -> bf16 conversion of x and Wq/Wk/Wv
//   Kernel 1: fused QKV projection (pure bf16 operands -> bf16 Q,K + V^T)
//   Kernel 2: flash-style attention (streaming softmax, WMMA scores + AV,
//             async LDS->global epilogue stores via ASYNCcnt DMA path)
// Workspace: Q/K/V^T (50.3 MB) + xh (16.8 MB) + Wh (6.3 MB) = 73.4 MB of d_ws.
// ---------------------------------------------------------------------------

#define BB 4
#define SS 2048
#define DD 1024
#define SCALE 0.03125f   // 1/sqrt(1024)

typedef float   f32x4  __attribute__((ext_vector_type(4)));
typedef float   v8f    __attribute__((ext_vector_type(8)));
typedef __bf16  bf16x8 __attribute__((ext_vector_type(8)));
typedef __bf16  v16bf  __attribute__((ext_vector_type(16)));

// Exact parameter type of the async-store builtin (from compiler diagnostic):
// int __attribute__((vector_size(16))) in addrspace(1) / addrspace(3).
typedef int v4i_t __attribute__((vector_size(16)));
typedef __attribute__((address_space(1))) v4i_t* g1_v4i_p;
typedef __attribute__((address_space(3))) v4i_t* l3_v4i_p;

// ---- WMMA wrapper: D = A(16x32 bf16) * B(32x16 bf16) + C(16x16 f32) --------
__device__ __forceinline__ v8f wmma_bf16(v16bf a, v16bf b, v8f c) {
  return __builtin_amdgcn_wmma_f32_16x16x32_bf16(
      /*neg_a=*/false, a, /*neg_b=*/false, b,
      /*c_mod=*/(short)0, c, /*reuse_a=*/false, /*reuse_b=*/false);
}

// ---- Fragment loader -------------------------------------------------------
// 16-bit A/B fragment, row-major source, contiguous K per lane:
//   lane = {row = lane&15, khalf = lane>>4}; K chunks [kh*8..+7], [16+kh*8..+7]
__device__ __forceinline__ v16bf frag_ld_bf16(const __bf16* tile, int ld, int lane) {
  const int r = lane & 15, h = (lane >> 4) & 1;
  const __bf16* p = tile + (size_t)r * ld + h * 8;
  bf16x8 c0 = *(const bf16x8*)(p);
  bf16x8 c1 = *(const bf16x8*)(p + 16);
  v16bf o;
#pragma unroll
  for (int i = 0; i < 8; ++i) { o[i] = c0[i]; o[i + 8] = c1[i]; }
  return o;
}

// ---------------------------------------------------------------------------
// Kernel 0: vectorized f32 -> bf16 conversion (8 elements / thread / step).
// ---------------------------------------------------------------------------
__global__ __launch_bounds__(256) void cvt_bf16_kernel(
    const float* __restrict__ src, __bf16* __restrict__ dst, int n8) {
  for (int i = blockIdx.x * blockDim.x + threadIdx.x; i < n8;
       i += gridDim.x * blockDim.x) {
    const float* p = src + (size_t)i * 8;
    f32x4 a = *(const f32x4*)(p);
    f32x4 b = *(const f32x4*)(p + 4);
    bf16x8 o;
#pragma unroll
    for (int j = 0; j < 4; ++j) { o[j] = (__bf16)a[j]; o[j + 4] = (__bf16)b[j]; }
    *(bf16x8*)(dst + (size_t)i * 8) = o;
  }
}

// ---------------------------------------------------------------------------
// Kernel 1: fused QKV projection, all-bf16 operands.
//   Q[bs,e] = sum_d x[bs,d] * Wq[e,d] + bq[e]   (same for K, V)
// Wave tile: 16 (M over bs) x 64 (N over e). 24576 wave-jobs total.
// Q,K stored bf16 row-major [B*S, D]; V stored transposed per batch [D, S].
// ---------------------------------------------------------------------------
__global__ __launch_bounds__(256) void qkv_kernel(
    const __bf16* __restrict__ xh, const __bf16* __restrict__ Wh,
    const float* __restrict__ bq, const float* __restrict__ bk,
    const float* __restrict__ bv,
    __bf16* __restrict__ Qh, __bf16* __restrict__ Kh, __bf16* __restrict__ Vt) {
  __shared__ float lds_t[8][16 * 17];   // per-wave transpose staging

  const int tid  = threadIdx.x;
  const int wave = tid >> 5;
  const int lane = tid & 31;

  const int job   = blockIdx.x * 8 + wave;
  const int mtile = job / 48;                 // 512 row tiles over B*S
  const int sub   = job % 48;
  const int which = sub >> 4;                 // 0=Q 1=K 2=V
  const int n0    = (sub & 15) * 64;          // output-column tile base
  const int row0  = mtile * 16;               // flat bs row base

  const __bf16* W    = Wh + (size_t)which * DD * DD;
  const float*  bias = (which == 0) ? bq : (which == 1) ? bk : bv;

  v8f acc[4] = {};
  const __bf16* xrow = xh + (size_t)row0 * DD;

  for (int k = 0; k < DD; k += 32) {
    if (k + 128 < DD)
      __builtin_prefetch(xrow + (size_t)(lane & 15) * DD + k + 128, 0, 1);
    v16bf a = frag_ld_bf16(xrow + k, DD, lane);
#pragma unroll
    for (int f = 0; f < 4; ++f) {
      v16bf bfrag = frag_ld_bf16(W + (size_t)(n0 + f * 16) * DD + k, DD, lane);
      acc[f] = wmma_bf16(a, bfrag, acc[f]);
    }
  }

  const int ncol = lane & 15;
  const int half = lane >> 4;
#pragma unroll
  for (int f = 0; f < 4; ++f) {
    float bv_ = bias[n0 + f * 16 + ncol];
#pragma unroll
    for (int v = 0; v < 8; ++v) acc[f][v] += bv_;
  }

  if (which == 2) {
    // Direct transposed store: C layout lane=N(e), VGPR=M(t) -> contiguous t.
    const int b   = row0 / SS;
    const int t0l = row0 % SS;
    __bf16* Vb = Vt + (size_t)b * DD * SS;
#pragma unroll
    for (int f = 0; f < 4; ++f) {
      const int e = n0 + f * 16 + ncol;
      bf16x8 o;
#pragma unroll
      for (int v = 0; v < 8; ++v) o[v] = (__bf16)acc[f][v];
      *(bf16x8*)(Vb + (size_t)e * SS + t0l + 8 * half) = o;
    }
  } else {
    // LDS transpose -> coalesced row-major bf16 stores.
    __bf16* dst = (which == 0) ? Qh : Kh;
    float* tb = &lds_t[wave][0];
#pragma unroll
    for (int f = 0; f < 4; ++f) {
#pragma unroll
      for (int v = 0; v < 8; ++v)
        tb[(v + 8 * half) * 17 + ncol] = acc[f][v];
      asm volatile("s_wait_dscnt 0x0" ::: "memory");
      bf16x8 o;
#pragma unroll
      for (int j = 0; j < 8; ++j)
        o[j] = (__bf16)tb[ncol * 17 + half * 8 + j];
      *(bf16x8*)(dst + (size_t)(row0 + ncol) * DD + n0 + f * 16 + half * 8) = o;
      asm volatile("s_wait_dscnt 0x0" ::: "memory");
    }
  }
}

// ---------------------------------------------------------------------------
// Kernel 2: streaming-softmax attention.
// One block (8 waves) per (batch, 16-query tile). Wave w owns e-slice
// [w*128, w*128+128) of the score contraction AND output columns
// [w*128, w*128+128). Per 32-key step: partial score WMMAs -> LDS tree
// reduce -> online softmax (16 lanes) -> P(bf16) -> AV WMMAs against V^T.
// ---------------------------------------------------------------------------
#define TP 20   // transpose-buffer pitch in floats (80B rows, 16B aligned)

__global__ __launch_bounds__(256) void attn_kernel(
    const __bf16* __restrict__ Qh, const __bf16* __restrict__ Kh,
    const __bf16* __restrict__ Vt, float* __restrict__ out) {
  __shared__ float  spart[8][16 * 32];  // per-wave score partials
  __shared__ __bf16 pt[16 * 32];        // probability tile (A for AV)
  __shared__ float  mrow[16], lrow[16], srow[16];
  __shared__ float  tbuf[8][16 * TP];   // per-wave output transpose

  const int tid  = threadIdx.x;
  const int wave = tid >> 5;
  const int lane = tid & 31;
  const int b    = blockIdx.x >> 7;          // S/16 = 128 tiles per batch
  const int s0   = (blockIdx.x & 127) << 4;
  const size_t bs0 = (size_t)b * SS + s0;

  if (tid < 16) { mrow[tid] = -3.0e38f; lrow[tid] = 0.0f; }
  __syncthreads();

  const int ew   = wave * 128;   // e-slice for score contraction
  const int oc0  = wave * 128;   // output-column slice
  const int n    = lane & 15;
  const int half = lane >> 4;
  const int mb   = 8 * half;

  const __bf16* Kb = Kh + (size_t)b * SS * DD;
  const __bf16* Vb = Vt + (size_t)b * DD * SS;

  // Q fragments for this wave's e-slice: resident for whole kernel.
  v16bf qf[4];
#pragma unroll
  for (int i = 0; i < 4; ++i)
    qf[i] = frag_ld_bf16(Qh + bs0 * DD + ew + i * 32, DD, lane);

  v8f oacc[8] = {};

  for (int t0 = 0; t0 < SS; t0 += 32) {
    // ---- partial scores: 2 key tiles x 4 K-steps of 32 ----
#pragma unroll
    for (int tt = 0; tt < 2; ++tt) {
      v8f s = {};
#pragma unroll
      for (int i = 0; i < 4; ++i) {
        v16bf kf = frag_ld_bf16(Kb + (size_t)(t0 + tt * 16) * DD + ew + i * 32,
                                DD, lane);
        s = wmma_bf16(qf[i], kf, s);
      }
#pragma unroll
      for (int v = 0; v < 8; ++v)
        spart[wave][(mb + v) * 32 + tt * 16 + n] = s[v];
    }
    __syncthreads();

    // ---- deterministic reduce + online softmax (16 lanes) ----
    if (tid < 16) {
      const int r = tid;
      float mo = mrow[r], mx = mo;
      float sv[32];
#pragma unroll
      for (int j = 0; j < 32; ++j) {
        float a = 0.0f;
#pragma unroll
        for (int w = 0; w < 8; ++w) a += spart[w][r * 32 + j];
        sv[j] = a * SCALE;
        mx = fmaxf(mx, sv[j]);
      }
      float sum = 0.0f;
#pragma unroll
      for (int j = 0; j < 32; ++j) {
        float p = __expf(sv[j] - mx);
        sum += p;
        pt[r * 32 + j] = (__bf16)p;
      }
      float corr = __expf(mo - mx);
      mrow[r] = mx;
      lrow[r] = lrow[r] * corr + sum;
      srow[r] = corr;
    }
    __syncthreads();

    // ---- rescale accumulators, then AV: P(16x32) @ V^T-slices ----
    float cf[8];
#pragma unroll
    for (int v = 0; v < 8; ++v) cf[v] = srow[mb + v];

    v16bf pf = frag_ld_bf16(&pt[0], 32, lane);
#pragma unroll
    for (int g = 0; g < 8; ++g) {
#pragma unroll
      for (int v = 0; v < 8; ++v) oacc[g][v] *= cf[v];
      v16bf vf = frag_ld_bf16(Vb + (size_t)(oc0 + g * 16) * SS + t0, SS, lane);
      oacc[g] = wmma_bf16(pf, vf, oacc[g]);
    }
    __syncthreads();
  }

  // ---- epilogue: 1/l normalize, LDS transpose, coalesced f32 stores ----
  float lf[8];
#pragma unroll
  for (int v = 0; v < 8; ++v) lf[v] = 1.0f / lrow[mb + v];

  float* tb = &tbuf[wave][0];
#pragma unroll
  for (int g = 0; g < 8; ++g) {
#pragma unroll
    for (int v = 0; v < 8; ++v)
      tb[(mb + v) * TP + n] = oacc[g][v] * lf[v];
    asm volatile("s_wait_dscnt 0x0" ::: "memory");

    float* op = out + (bs0 + n) * DD + oc0 + g * 16 + half * 8;
    float* lp = tb + n * TP + half * 8;   // 16B-aligned LDS source

#if defined(__gfx1250__) && __has_builtin(__builtin_amdgcn_global_store_async_from_lds_b128)
    // ASYNCcnt-tracked LDS -> global DMA store (two 16B beats per lane).
    __builtin_amdgcn_global_store_async_from_lds_b128(
        (g1_v4i_p)(unsigned long long)op,
        (l3_v4i_p)(unsigned int)(unsigned long long)lp,
        0, 0);
    __builtin_amdgcn_global_store_async_from_lds_b128(
        (g1_v4i_p)(unsigned long long)(op + 4),
        (l3_v4i_p)(unsigned int)(unsigned long long)(lp + 4),
        0, 0);
#if __has_builtin(__builtin_amdgcn_s_wait_asynccnt)
    __builtin_amdgcn_s_wait_asynccnt(0);   // WAR: tbuf reused next iteration
#else
    asm volatile("s_wait_asynccnt 0x0" ::: "memory");
#endif
#else
    f32x4 o0, o1;
#pragma unroll
    for (int j = 0; j < 4; ++j) { o0[j] = lp[j]; o1[j] = lp[4 + j]; }
    *(f32x4*)(op)     = o0;
    *(f32x4*)(op + 4) = o1;
    asm volatile("s_wait_dscnt 0x0" ::: "memory");
#endif
  }
}

// ---------------------------------------------------------------------------
extern "C" void kernel_launch(void* const* d_in, const int* in_sizes, int n_in,
                              void* d_out, int out_size, void* d_ws, size_t ws_size,
                              hipStream_t stream) {
  const float* x  = (const float*)d_in[0];
  const float* Wq = (const float*)d_in[1];
  const float* bq = (const float*)d_in[2];
  const float* Wk = (const float*)d_in[3];
  const float* bk = (const float*)d_in[4];
  const float* Wv = (const float*)d_in[5];
  const float* bv = (const float*)d_in[6];
  float* out = (float*)d_out;

  const size_t nElem = (size_t)BB * SS * DD;        // 8,388,608
  const size_t nW    = (size_t)DD * DD;             // 1,048,576
  __bf16* Qh = (__bf16*)d_ws;                       // [B*S, D] bf16
  __bf16* Kh = Qh + nElem;                          // [B*S, D] bf16
  __bf16* Vt = Kh + nElem;                          // per-batch [D, S] bf16
  __bf16* xh = Vt + nElem;                          // [B*S, D] bf16
  __bf16* Wh = xh + nElem;                          // [3, D, D] bf16
  // total ws use: (3*nElem + nElem + 3*nW) * 2 = 73,400,320 bytes

  // --- one-shot bf16 conversion of operands ---
  {
    int n8 = (int)(nElem / 8);
    cvt_bf16_kernel<<<(n8 + 255) / 256, 256, 0, stream>>>(x, xh, n8);
    int w8 = (int)(nW / 8);
    cvt_bf16_kernel<<<(w8 + 255) / 256, 256, 0, stream>>>(Wq, Wh + 0 * nW, w8);
    cvt_bf16_kernel<<<(w8 + 255) / 256, 256, 0, stream>>>(Wk, Wh + 1 * nW, w8);
    cvt_bf16_kernel<<<(w8 + 255) / 256, 256, 0, stream>>>(Wv, Wh + 2 * nW, w8);
  }

  // 512 M-tiles * 48 (3 outputs x 16 col-tiles) = 24576 wave jobs / 8 per block
  qkv_kernel<<<3072, 256, 0, stream>>>(xh, Wh, bq, bk, bv, Qh, Kh, Vt);
  // one block per (batch, 16-query tile)
  attn_kernel<<<BB * (SS / 16), 256, 0, stream>>>(Qh, Kh, Vt, out);
}
